// TransformerBlock_19078244728946
// MI455X (gfx1250) — matrix operand order repaired and checked by reading, not compiled, not run on hardware
//
#include <hip/hip_runtime.h>
#include <hip/hip_bf16.h>

// ---------------------------------------------------------------------------
// Transformer block for MI455X (gfx1250): bf16 WMMA everywhere, f32 accum.
// Async global->LDS staging (ASYNCcnt) with double buffering in the GEMMs.
// ---------------------------------------------------------------------------

typedef __bf16 bf16_t;
typedef __bf16 v16bf __attribute__((ext_vector_type(16)));
typedef float  v8f   __attribute__((ext_vector_type(8)));

#define B_  8
#define T_  1024
#define D_  1024
#define H_  16
#define HD_ 64
#define DF_ 4096
#define EPS_ 1e-5f

#ifndef __has_builtin
#define __has_builtin(x) 0
#endif
#if __has_builtin(__builtin_amdgcn_global_load_async_to_lds_b128) && \
    __has_builtin(__builtin_amdgcn_s_wait_asynccnt)
#define HAVE_ASYNC_LDS 1
#else
#define HAVE_ASYNC_LDS 0
#endif

// Pointer types matching the builtin signature reported by clang:
//   param0: 'int __vector_size__(16) __device__ *'  (global, AS1)
//   param1: LDS destination (AS3)
typedef int i32x4_ __attribute__((__vector_size__(4 * sizeof(int))));
typedef __attribute__((address_space(1))) i32x4_* as1_i32x4p;
typedef __attribute__((address_space(3))) i32x4_* as3_i32x4p;

// 16B global -> LDS copy; async (ASYNCcnt-tracked) when the toolchain has it.
static __device__ __forceinline__ void async_copy16(const bf16_t* g, bf16_t* l) {
#if HAVE_ASYNC_LDS
    __builtin_amdgcn_global_load_async_to_lds_b128((as1_i32x4p)g, (as3_i32x4p)l, 0, 0);
#else
    *(uint4*)l = *(const uint4*)g;
#endif
}
static __device__ __forceinline__ void async_wait_le4() {
#if HAVE_ASYNC_LDS
    __builtin_amdgcn_s_wait_asynccnt(4);
#endif
}
static __device__ __forceinline__ void async_wait_0() {
#if HAVE_ASYNC_LDS
    __builtin_amdgcn_s_wait_asynccnt(0);
#endif
}

// ---------------- WMMA helpers (layouts per CDNA5 ISA 7.12.2) ----------------

// A-matrix 16x32 bf16 fragment. lane L: row m = L&15; half = L>>4.
// elements e<8  -> k = half*8 + e ; e>=8 -> k = 16 + half*8 + (e-8)
static __device__ __forceinline__ v16bf load_a_frag(const bf16_t* base, int ld, int lane) {
    const int m = lane & 15, half = lane >> 4;
    const bf16_t* r = base + (size_t)m * ld + half * 8;
    v16bf f;
#pragma unroll
    for (int e = 0; e < 8; ++e) f[e] = r[e];
#pragma unroll
    for (int e = 0; e < 8; ++e) f[8 + e] = r[16 + e];
    return f;
}

// B-matrix 32x16 bf16 fragment, source stored row-major as [N][K] (so that
// B[k][n] = src[n][k]). lane L: n = L&15; e<8 -> k = (L>>4)*8 + e ; e>=8 -> k = 16 + (L>>4)*8 + (e-8).
static __device__ __forceinline__ v16bf load_b_frag(const bf16_t* base, int ld, int lane) {
    const int n = lane & 15, half = lane >> 4;
    const bf16_t* r = base + (size_t)n * ld + half * 8;
    v16bf f;
#pragma unroll
    for (int e = 0; e < 8; ++e) f[e] = r[e];
#pragma unroll
    for (int e = 0; e < 8; ++e) f[8 + e] = r[16 + e];
    return f;
}

// C/D 16x16 f32: VGPR r, lane L: row = r + 8*(L>>4), col = L&15.
static __device__ __forceinline__ v8f wmma_bf16(v16bf a, v16bf b, v8f c) {
    return __builtin_amdgcn_wmma_f32_16x16x32_bf16(false, a, false, b, (short)0, c,
                                                   false, false);
}

// ------------------------------- conversion ---------------------------------

__global__ void cvt_f32_bf16(const float* __restrict__ s, bf16_t* __restrict__ d, int n) {
    for (int i = blockIdx.x * blockDim.x + threadIdx.x; i < n; i += gridDim.x * blockDim.x)
        d[i] = (bf16_t)s[i];
}

// ------------------------- per-head QKV projections --------------------------
// One wave computes a 16x64 strip of one head projection: X[16x64] @ W[64x64]^T
// 3072 blocks * 8 waves = 3 matrices * 8 b * 16 h * 64 m-tiles.

__global__ void qkv_proj(const bf16_t* __restrict__ qbf, const bf16_t* __restrict__ kbf,
                         const bf16_t* __restrict__ vbf,
                         const bf16_t* __restrict__ Wq, const bf16_t* __restrict__ Wk,
                         const bf16_t* __restrict__ Wv,
                         bf16_t* __restrict__ qp, bf16_t* __restrict__ kp,
                         bf16_t* __restrict__ vp) {
    const int tid = threadIdx.x, lane = tid & 31, wid = tid >> 5;
    const int wg = blockIdx.x * 8 + wid;
    const int mtile = wg & 63;
    const int h     = (wg >> 6) & 15;
    const int b     = (wg >> 10) & 7;
    const int mat   = wg >> 13;             // 0=q, 1=k, 2=v

    const bf16_t* X; const bf16_t* W; bf16_t* O;
    if (mat == 0)      { X = qbf; W = Wq; O = qp; }
    else if (mat == 1) { X = kbf; W = Wk; O = kp; }
    else               { X = vbf; W = Wv; O = vp; }

    const int half = lane >> 4, l15 = lane & 15;
    const bf16_t* abase = X + ((size_t)(b * T_ + mtile * 16)) * D_ + h * HD_;
    v16bf a0 = load_a_frag(abase,      D_, lane);   // k = 0..31
    v16bf a1 = load_a_frag(abase + 32, D_, lane);   // k = 32..63

#pragma unroll
    for (int nt = 0; nt < 4; ++nt) {
        v16bf b0 = load_b_frag(W + (nt * 16) * 64,      64, lane);
        v16bf b1 = load_b_frag(W + (nt * 16) * 64 + 32, 64, lane);
        v8f cc = {};
        cc = wmma_bf16(a0, b0, cc);
        cc = wmma_bf16(a1, b1, cc);
#pragma unroll
        for (int r = 0; r < 8; ++r) {
            const int row = mtile * 16 + r + 8 * half;
            O[((size_t)(b * T_ + row)) * D_ + h * HD_ + nt * 16 + l15] = (bf16_t)cc[r];
        }
    }
}

// ------------------------------ flash attention ------------------------------
// Block = 256 threads (8 waves) handles (b, h, 128 query rows). Each wave owns
// 16 query rows. Keys processed in tiles of 64: K tile staged (async) to LDS,
// V tile staged transposed so V B-fragments are contiguous LDS reads.

__global__ void flash_attn(const bf16_t* __restrict__ qp, const bf16_t* __restrict__ kp,
                           const bf16_t* __restrict__ vp, const int* __restrict__ mask,
                           bf16_t* __restrict__ attn) {
    __shared__ bf16_t Kld[64 * 64];     // [key][d]
    __shared__ bf16_t Vt[64 * 64];      // [d(e)][key]  (transposed)
    __shared__ bf16_t Pb[8][16 * 64];   // per-wave P staging

    const int tid = threadIdx.x, lane = tid & 31, wid = tid >> 5;
    const int half = lane >> 4, l15 = lane & 15;
    const int bx = blockIdx.x;
    const int qblk = bx & 7, h = (bx >> 3) & 15, b = bx >> 7;
    const int qbase = qblk * 128 + wid * 16;

    // Q fragments (persistent)
    const bf16_t* qbase_p = qp + ((size_t)(b * T_ + qbase)) * D_ + h * HD_;
    v16bf a0 = load_a_frag(qbase_p,      D_, lane);
    v16bf a1 = load_a_frag(qbase_p + 32, D_, lane);

    float m_i[8], l_i[8];
    v8f acc[4] = {};
#pragma unroll
    for (int r = 0; r < 8; ++r) { m_i[r] = -3.0e38f; l_i[r] = 0.f; }

    const float sc = 0.03125f;  // 1/sqrt(D)

    for (int kb = 0; kb < T_ / 64; ++kb) {
        // ---- stage K tile (async) and transposed V tile ----
        {
            const bf16_t* ksrc = kp + ((size_t)(b * T_ + kb * 64)) * D_ + h * HD_;
            const bf16_t* vsrc = vp + ((size_t)(b * T_ + kb * 64)) * D_ + h * HD_;
#pragma unroll
            for (int c2 = 0; c2 < 2; ++c2) {
                const int idx = tid * 2 + c2;          // 0..511 16B chunks
                const int krow = idx >> 3, sub = idx & 7;
                async_copy16(ksrc + (size_t)krow * D_ + sub * 8,
                             &Kld[krow * 64 + sub * 8]);
            }
#pragma unroll
            for (int c2 = 0; c2 < 2; ++c2) {
                const int idx = tid * 2 + c2;
                const int vrow = idx >> 3, sub = idx & 7;
                uint4 u = *(const uint4*)(vsrc + (size_t)vrow * D_ + sub * 8);
                const bf16_t* pe = (const bf16_t*)&u;
#pragma unroll
                for (int j = 0; j < 8; ++j) Vt[(sub * 8 + j) * 64 + vrow] = pe[j];
            }
        }
        async_wait_0();
        __syncthreads();

        // ---- S = Q @ K^T (16 x 64) ----
        v8f s[4];
#pragma unroll
        for (int nt = 0; nt < 4; ++nt) {
            v16bf b0 = load_b_frag(Kld + (nt * 16) * 64,      64, lane);
            v16bf b1 = load_b_frag(Kld + (nt * 16) * 64 + 32, 64, lane);
            v8f z = {};
            z = wmma_bf16(a0, b0, z);
            z = wmma_bf16(a1, b1, z);
            s[nt] = z;
        }

        // ---- scale + mask + online softmax ----
        float mk[4];
#pragma unroll
        for (int nt = 0; nt < 4; ++nt)
            mk[nt] = (mask[b * T_ + kb * 64 + nt * 16 + l15] == 0) ? -1e20f : 0.f;

#pragma unroll
        for (int r = 0; r < 8; ++r) {
            float v = -3.4e38f;
#pragma unroll
            for (int nt = 0; nt < 4; ++nt) {
                float x = s[nt][r] * sc + mk[nt];
                s[nt][r] = x;
                v = fmaxf(v, x);
            }
            // reduce across the 16 lanes of this half (C cols stripe 16 lanes)
            v = fmaxf(v, __shfl_xor(v, 1));
            v = fmaxf(v, __shfl_xor(v, 2));
            v = fmaxf(v, __shfl_xor(v, 4));
            v = fmaxf(v, __shfl_xor(v, 8));
            const float mnew = fmaxf(m_i[r], v);
            const float scl  = __expf(m_i[r] - mnew);
            m_i[r] = mnew;
            float sum = 0.f;
#pragma unroll
            for (int nt = 0; nt < 4; ++nt) {
                float p = __expf(s[nt][r] - mnew);
                s[nt][r] = p;
                sum += p;
            }
            sum += __shfl_xor(sum, 1);
            sum += __shfl_xor(sum, 2);
            sum += __shfl_xor(sum, 4);
            sum += __shfl_xor(sum, 8);
            l_i[r] = l_i[r] * scl + sum;
#pragma unroll
            for (int et = 0; et < 4; ++et) acc[et][r] = acc[et][r] * scl;
        }

        // ---- relayout P (C-frag -> A-frag) through per-wave LDS tile ----
        bf16_t* pw = &Pb[wid][0];
#pragma unroll
        for (int nt = 0; nt < 4; ++nt)
#pragma unroll
            for (int r = 0; r < 8; ++r)
                pw[(r + 8 * half) * 64 + nt * 16 + l15] = (bf16_t)s[nt][r];

        v16bf p0 = load_a_frag(pw,      64, lane);   // keys 0..31
        v16bf p1 = load_a_frag(pw + 32, 64, lane);   // keys 32..63

        // ---- O += P @ V ----
#pragma unroll
        for (int et = 0; et < 4; ++et) {
            v16bf vb0 = load_b_frag(Vt + (et * 16) * 64,      64, lane);
            v16bf vb1 = load_b_frag(Vt + (et * 16) * 64 + 32, 64, lane);
            acc[et] = wmma_bf16(p0, vb0, acc[et]);
            acc[et] = wmma_bf16(p1, vb1, acc[et]);
        }
        __syncthreads();
    }

    // ---- epilogue: O /= l, store bf16 ----
#pragma unroll
    for (int r = 0; r < 8; ++r) {
        const float inv = 1.0f / l_i[r];
        const int row = qbase + r + 8 * half;
#pragma unroll
        for (int et = 0; et < 4; ++et)
            attn[((size_t)(b * T_ + row)) * D_ + h * HD_ + et * 16 + l15] =
                (bf16_t)(acc[et][r] * inv);
    }
}

// ------------------------------ generic GEMM ---------------------------------
// out = A[MxK](bf16) @ W[NxK](bf16)^T + bias.
// Block tile 128x128, 8 waves of 32x64, K-step 32, double-buffered async LDS
// staging. RELU_BF16: relu + bf16 out, else f32 out.

template <int RELU_BF16>
__global__ void gemm_wmma(const bf16_t* __restrict__ A, const bf16_t* __restrict__ W,
                          const float* __restrict__ bias, float* __restrict__ outF,
                          bf16_t* __restrict__ outB, int M, int N, int K) {
    __shared__ bf16_t As[2][128 * 32];
    __shared__ bf16_t Ws[2][128 * 32];
    const int tid = threadIdx.x, lane = tid & 31, wid = tid >> 5;
    const int half = lane >> 4, l15 = lane & 15;
    const int m0 = blockIdx.y * 128, n0 = blockIdx.x * 128;
    const int wm = (wid >> 1) * 32, wn = (wid & 1) * 64;

    const int ksteps = K >> 5;

    auto stage = [&](int kk, int bufi) {
        const int k0 = kk * 32;
#pragma unroll
        for (int i = 0; i < 2; ++i) {
            const int idx = tid + 256 * i;     // 0..511 16B chunks (128 rows x 4)
            const int row = idx >> 2, sub = idx & 3;
            async_copy16(A + ((size_t)(m0 + row)) * K + k0 + sub * 8,
                         &As[bufi][row * 32 + sub * 8]);
            async_copy16(W + ((size_t)(n0 + row)) * K + k0 + sub * 8,
                         &Ws[bufi][row * 32 + sub * 8]);
        }
    };

    v8f c[2][4] = {};
    stage(0, 0);
    for (int kk = 0; kk < ksteps; ++kk) {
        const int cur = kk & 1;
        if (kk + 1 < ksteps) {
            stage(kk + 1, cur ^ 1);
            async_wait_le4();   // 4 in flight for next stage; current stage done
        } else {
            async_wait_0();
        }
        __syncthreads();

        v16bf a0 = load_a_frag(&As[cur][(wm + 0) * 32],  32, lane);
        v16bf a1 = load_a_frag(&As[cur][(wm + 16) * 32], 32, lane);
#pragma unroll
        for (int nt = 0; nt < 4; ++nt) {
            v16bf bf = load_b_frag(&Ws[cur][(wn + nt * 16) * 32], 32, lane);
            c[0][nt] = wmma_bf16(a0, bf, c[0][nt]);
            c[1][nt] = wmma_bf16(a1, bf, c[1][nt]);
        }
        __syncthreads();
    }

#pragma unroll
    for (int mt = 0; mt < 2; ++mt)
#pragma unroll
        for (int nt = 0; nt < 4; ++nt) {
            const int colg = n0 + wn + nt * 16 + l15;
            const float bia = bias[colg];
#pragma unroll
            for (int r = 0; r < 8; ++r) {
                const int rowg = m0 + wm + mt * 16 + r + 8 * half;
                float vv = c[mt][nt][r] + bia;
                if (RELU_BF16) {
                    vv = fmaxf(vv, 0.f);
                    outB[(size_t)rowg * N + colg] = (bf16_t)vv;
                } else {
                    outF[(size_t)rowg * N + colg] = vv;
                }
            }
        }
}

// --------------------------- residual + LayerNorm ----------------------------

__global__ void add_ln_kernel(const float* __restrict__ y, const float* __restrict__ res,
                              const float* __restrict__ g, const float* __restrict__ be,
                              float* __restrict__ outF, bf16_t* __restrict__ outB) {
    const int row = blockIdx.x;
    const float* yr = y + (size_t)row * D_;
    const float* rr = res + (size_t)row * D_;
    const int tid = threadIdx.x;

    float v[4];
    float s = 0.f, s2 = 0.f;
#pragma unroll
    for (int i = 0; i < 4; ++i) {
        const float x = yr[tid + i * 256] + rr[tid + i * 256];
        v[i] = x;
        s += x;
        s2 += x * x;
    }
#pragma unroll
    for (int m = 16; m >= 1; m >>= 1) {
        s += __shfl_xor(s, m);
        s2 += __shfl_xor(s2, m);
    }
    __shared__ float sA[8], sB[8];
    const int wid = tid >> 5, lane = tid & 31;
    if (lane == 0) { sA[wid] = s; sB[wid] = s2; }
    __syncthreads();
    if (tid == 0) {
        float a = 0.f, b2 = 0.f;
#pragma unroll
        for (int i = 0; i < 8; ++i) { a += sA[i]; b2 += sB[i]; }
        sA[0] = a; sB[0] = b2;
    }
    __syncthreads();
    const float mu = sA[0] * (1.0f / D_);
    const float var = sB[0] * (1.0f / D_) - mu * mu;
    const float rstd = rsqrtf(var + EPS_);
#pragma unroll
    for (int i = 0; i < 4; ++i) {
        const int cidx = tid + i * 256;
        const float o = (v[i] - mu) * rstd * g[cidx] + be[cidx];
        outF[(size_t)row * D_ + cidx] = o;
        if (outB) outB[(size_t)row * D_ + cidx] = (bf16_t)o;
    }
}

// --------------------------------- driver ------------------------------------

extern "C" void kernel_launch(void* const* d_in, const int* in_sizes, int n_in,
                              void* d_out, int out_size, void* d_ws, size_t ws_size,
                              hipStream_t stream) {
    const float* value = (const float*)d_in[0];
    const float* key   = (const float*)d_in[1];
    const float* query = (const float*)d_in[2];
    const int*   mask  = (const int*)d_in[3];
    const float* Wk    = (const float*)d_in[4];
    const float* Wq    = (const float*)d_in[5];
    const float* Wv    = (const float*)d_in[6];
    const float* Wu    = (const float*)d_in[7];
    const float* bu    = (const float*)d_in[8];
    const float* ln1g  = (const float*)d_in[9];
    const float* ln1b  = (const float*)d_in[10];
    const float* ln2g  = (const float*)d_in[11];
    const float* ln2b  = (const float*)d_in[12];
    const float* W1    = (const float*)d_in[13];
    const float* b1    = (const float*)d_in[14];
    const float* W2    = (const float*)d_in[15];
    const float* b2    = (const float*)d_in[16];
    float* out = (float*)d_out;

    char* ws = (char*)d_ws;
    const size_t MB = 1024ull * 1024ull;
    // phase A (pre-attention)
    bf16_t* value_bf = (bf16_t*)(ws + 0);        // 16MB
    bf16_t* key_bf   = (bf16_t*)(ws + 16 * MB);  // 16MB
    bf16_t* query_bf = (bf16_t*)(ws + 32 * MB);  // 16MB
    // phase B: FFN hidden reuses the (dead) region above
    bf16_t* h_bf     = (bf16_t*)(ws + 0);        // 64MB
    bf16_t* qp       = (bf16_t*)(ws + 64 * MB);  // 16MB
    bf16_t* kp       = (bf16_t*)(ws + 80 * MB);  // 16MB
    bf16_t* vp       = (bf16_t*)(ws + 96 * MB);  // 16MB
    bf16_t* attn_bf  = (bf16_t*)(ws + 112 * MB); // 16MB
    float*  ybuf     = (float*)(ws + 128 * MB);  // 32MB (reused for FFN2 out)
    float*  x_f      = (float*)(ws + 160 * MB);  // 32MB
    bf16_t* x_bf     = (bf16_t*)(ws + 192 * MB); // 16MB
    bf16_t* Wk_bf    = (bf16_t*)(ws + 208 * MB);
    bf16_t* Wq_bf    = (bf16_t*)(ws + 208 * MB + 8192);
    bf16_t* Wv_bf    = (bf16_t*)(ws + 208 * MB + 16384);
    bf16_t* Wu_bf    = (bf16_t*)(ws + 209 * MB); // 2MB
    bf16_t* W1_bf    = (bf16_t*)(ws + 212 * MB); // 8MB
    bf16_t* W2_bf    = (bf16_t*)(ws + 220 * MB); // 8MB  -> total ~228MB

    const int nBTD = B_ * T_ * D_;
    cvt_f32_bf16<<<2048, 256, 0, stream>>>(value, value_bf, nBTD);
    cvt_f32_bf16<<<2048, 256, 0, stream>>>(key, key_bf, nBTD);
    cvt_f32_bf16<<<2048, 256, 0, stream>>>(query, query_bf, nBTD);
    cvt_f32_bf16<<<16, 256, 0, stream>>>(Wk, Wk_bf, HD_ * HD_);
    cvt_f32_bf16<<<16, 256, 0, stream>>>(Wq, Wq_bf, HD_ * HD_);
    cvt_f32_bf16<<<16, 256, 0, stream>>>(Wv, Wv_bf, HD_ * HD_);
    cvt_f32_bf16<<<2048, 256, 0, stream>>>(Wu, Wu_bf, D_ * D_);
    cvt_f32_bf16<<<2048, 256, 0, stream>>>(W1, W1_bf, DF_ * D_);
    cvt_f32_bf16<<<2048, 256, 0, stream>>>(W2, W2_bf, D_ * DF_);

    // per-head projections: 3*8*16*64 wave tasks / 8 waves per block
    qkv_proj<<<3072, 256, 0, stream>>>(query_bf, key_bf, value_bf,
                                       Wq_bf, Wk_bf, Wv_bf, qp, kp, vp);

    // attention: B * H * (T/128) blocks
    flash_attn<<<1024, 256, 0, stream>>>(qp, kp, vp, mask, attn_bf);

    const int M = B_ * T_;
    // attn_out @ Wu^T + bu
    gemm_wmma<0><<<dim3(D_ / 128, M / 128), 256, 0, stream>>>(
        attn_bf, Wu_bf, bu, ybuf, nullptr, M, D_, D_);
    // x = LN1(y + query)
    add_ln_kernel<<<M, 256, 0, stream>>>(ybuf, query, ln1g, ln1b, x_f, x_bf);
    // h = relu(x @ W1^T + b1)  (bf16)
    gemm_wmma<1><<<dim3(DF_ / 128, M / 128), 256, 0, stream>>>(
        x_bf, W1_bf, b1, nullptr, h_bf, M, DF_, D_);
    // z = h @ W2^T + b2
    gemm_wmma<0><<<dim3(D_ / 128, M / 128), 256, 0, stream>>>(
        h_bf, W2_bf, b2, ybuf, nullptr, M, D_, DF_);
    // out = LN2(z + x)
    add_ln_kernel<<<M, 256, 0, stream>>>(ybuf, x_f, ln2g, ln2b, out, nullptr);
}
